// GraphAttentionLayer_73340861546946
// MI455X (gfx1250) — matrix operand order repaired and checked by reading, not compiled, not run on hardware
//
#include <hip/hip_runtime.h>

typedef __attribute__((ext_vector_type(16))) _Float16 v16h;
typedef __attribute__((ext_vector_type(8)))  _Float16 v8h;
typedef __attribute__((ext_vector_type(2)))  _Float16 v2h;
typedef __attribute__((ext_vector_type(8)))  float    v8f;
typedef __attribute__((ext_vector_type(4)))  float    v4f;
typedef __attribute__((ext_vector_type(2)))  float    v2f;
typedef __attribute__((ext_vector_type(2)))  int      v2i;

constexpr int   kN    = 8192;
constexpr int   kFin  = 512;
constexpr int   kFout = 256;
constexpr float kAlpha  = 0.2f;
constexpr float kNegBig = -9e15f;

__device__ __forceinline__ float wave_max_f(float v) {
#pragma unroll
  for (int m = 16; m >= 1; m >>= 1) v = fmaxf(v, __shfl_xor(v, m, 32));
  return v;
}
__device__ __forceinline__ float wave_sum_f(float v) {
#pragma unroll
  for (int m = 16; m >= 1; m >>= 1) v += __shfl_xor(v, m, 32);
  return v;
}

// ---------------------------------------------------------------- kernel 0
// WT[n][k] = (f16) W[k][n]   (256 x 512)
__global__ void k_transpose_W(const float* __restrict__ W,
                              _Float16* __restrict__ WT) {
  int n = blockIdx.x;                       // 0..255
  for (int k = threadIdx.x; k < kFin; k += blockDim.x)
    WT[n * kFin + k] = (_Float16)W[k * kFout + n];
}

// ---------------------------------------------------------------- kernel 1
// Wh = h @ W via v_wmma_f32_16x16x32_f16.  Writes Wh (f32, row major) and
// WhT (f16, [256][8192], i.e. Wh transposed) for the attention B-fragments.
__global__ void __launch_bounds__(256)
k_wh(const float* __restrict__ h, const _Float16* __restrict__ WT,
     float* __restrict__ Wh, _Float16* __restrict__ WhT) {
  __shared__ alignas(32) _Float16 lds[8][16 * 17];
  const int wave = threadIdx.x >> 5;
  const int lane = threadIdx.x & 31;
  const int gw   = blockIdx.x * 8 + wave;       // 0..8191 tiles
  const int r0   = (gw >> 4) * 16;              // row tile
  const int n0   = (gw & 15) * 16;              // col tile
  const int rl   = lane & 15;
  const int hi   = lane >> 4;                   // 0 | 1

  v8f c = {};
  const float*    hrow = h  + (r0 + rl) * kFin;
  const _Float16* wrow = WT + (n0 + rl) * kFin;

  for (int k0 = 0; k0 < kFin; k0 += 32) {
    // A fragment: lane<16 holds K 0..15; lane>=16 holds K 16..31
    const int ka = k0 + hi * 16;
    v4f a0 = *(const v4f*)(hrow + ka);
    v4f a1 = *(const v4f*)(hrow + ka + 4);
    v4f a2 = *(const v4f*)(hrow + ka + 8);
    v4f a3 = *(const v4f*)(hrow + ka + 12);
    v16h a;
#pragma unroll
    for (int i = 0; i < 4; ++i) {
      a[i]      = (_Float16)a0[i];
      a[4 + i]  = (_Float16)a1[i];
      a[8 + i]  = (_Float16)a2[i];
      a[12 + i] = (_Float16)a3[i];
    }
    // B fragment: lane<16 -> K 0..15, lane>=16 -> K 16..31, contiguous in WT
    v16h b = *(const v16h*)(wrow + k0 + hi * 16);
    c = __builtin_amdgcn_wmma_f32_16x16x32_f16(false, a, false, b,
                                               (short)0, c, false, false);
  }

  // C layout: VGPR j -> row j (lanes 0-15) / row j+8 (lanes 16-31), col = lane%16
#pragma unroll
  for (int j = 0; j < 8; ++j) {
    int rr = j + hi * 8;
    Wh[(r0 + rr) * kFout + n0 + rl] = c[j];
    lds[wave][rr * 17 + rl] = (_Float16)c[j];
  }
  __syncthreads();
  // transposed store: lane handles column cc of the tile, rows hi*8..hi*8+7
  v8h t;
#pragma unroll
  for (int i = 0; i < 8; ++i) t[i] = lds[wave][(hi * 8 + i) * 17 + rl];
  *(v8h*)(WhT + (n0 + rl) * kN + r0 + hi * 8) = t;
}

// ---------------------------------------------------------------- kernel 2
// f1[r] = Wh[r,:].a[0:256],  f2[r] = Wh[r,:].a[256:512]
__global__ void k_f12(const float* __restrict__ Wh, const float* __restrict__ av,
                      float* __restrict__ f1, float* __restrict__ f2) {
  const int wave = threadIdx.x >> 5, lane = threadIdx.x & 31;
  const int row  = blockIdx.x * 8 + wave;
  const float* wr = Wh + row * kFout + lane * 8;
  float s1 = 0.f, s2 = 0.f;
#pragma unroll
  for (int i = 0; i < 8; ++i) {
    float w = wr[i];
    s1 += w * av[lane * 8 + i];
    s2 += w * av[kFout + lane * 8 + i];
  }
  s1 = wave_sum_f(s1);
  s2 = wave_sum_f(s2);
  if (lane == 0) { f1[row] = s1; f2[row] = s2; }
}

// ---------------------------------------------------------------- kernel 2b
__global__ void k_f2max(const float* __restrict__ f2, float* __restrict__ out) {
  __shared__ float red[8];
  const int lane = threadIdx.x & 31, wave = threadIdx.x >> 5;
  float m = -INFINITY;
  for (int i = threadIdx.x; i < kN; i += 256) m = fmaxf(m, f2[i]);
  m = wave_max_f(m);
  if (lane == 0) red[wave] = m;
  __syncthreads();
  if (threadIdx.x == 0) {
    float r = red[0];
#pragma unroll
    for (int i = 1; i < 8; ++i) r = fmaxf(r, red[i]);
    *out = r;
  }
}

// ---------------------------------------------------------------- kernel 3
// Fused masked-softmax + attention@Wh.  One block = 16 output rows.
// Wave w owns softmax lanes for local rows {w, w+8} (columns 2*lane, 2*lane+1
// of each 64-wide chunk) and output columns [32w, 32w+32).
// Streams adj exactly once (268 MB total -> HBM roofline floor).
constexpr int kPitch = 72;                  // P-row pitch in f16 (144 B): makes
                                            // ds_load_b128 A-frag reads bank-
                                            // conflict free (stride-4 coverage)
__global__ void __launch_bounds__(256)
k_attn(const int* __restrict__ adj, const float* __restrict__ f1,
       const float* __restrict__ f2, const float* __restrict__ f2maxp,
       const _Float16* __restrict__ WhT, float* __restrict__ out) {
  __shared__ alignas(32) _Float16 ldsP[2][16 * kPitch];  // double-buffered P
  __shared__ float ldsS[16];

  const int wave = threadIdx.x >> 5;
  const int lane = threadIdx.x & 31;
  const int r0   = blockIdx.x * 16;
  const int rA   = wave, rB = wave + 8;     // local rows for softmax work
  const int rl   = lane & 15;
  const int hi   = lane >> 4;
  const int n0   = wave * 32;

  // Safe per-row softmax bound: leaky is monotone, so
  //   max_j e_rj <= leaky(f1_r + max_j f2_j) = M_r   (tight to ~1 ulp of max)
  const float f2m = *f2maxp;
  const float f1A = f1[r0 + rA], f1B = f1[r0 + rB];
  float MA = f1A + f2m; MA = MA > 0.f ? MA : kAlpha * MA;
  float MB = f1B + f2m; MB = MB > 0.f ? MB : kAlpha * MB;

  float sA = 0.f, sB = 0.f;
  v8f acc0 = {}, acc1 = {};

  const int* adjA = adj + (r0 + rA) * kN + 2 * lane;
  const int* adjB = adj + (r0 + rB) * kN + 2 * lane;
  const _Float16* bpA = WhT + (n0 + rl) * kN + hi * 16;        // cols n0..
  const _Float16* bpB = bpA + 16 * kN;                         // cols n0+16..

  for (int cb = 0; cb < kN; cb += 64) {
    // ---- scores for 64 columns of rows rA,rB (b64 coalesced adj reads)
    const v2i aA = *(const v2i*)(adjA + cb);
    const v2i aB = *(const v2i*)(adjB + cb);
    const v2f f2v = *(const v2f*)(f2 + cb + 2 * lane);

    float eA0 = f1A + f2v[0]; eA0 = eA0 > 0.f ? eA0 : kAlpha * eA0;
    eA0 = aA[0] > 0 ? eA0 : kNegBig;
    float eA1 = f1A + f2v[1]; eA1 = eA1 > 0.f ? eA1 : kAlpha * eA1;
    eA1 = aA[1] > 0 ? eA1 : kNegBig;
    float eB0 = f1B + f2v[0]; eB0 = eB0 > 0.f ? eB0 : kAlpha * eB0;
    eB0 = aB[0] > 0 ? eB0 : kNegBig;
    float eB1 = f1B + f2v[1]; eB1 = eB1 > 0.f ? eB1 : kAlpha * eB1;
    eB1 = aB[1] > 0 ? eB1 : kNegBig;

    const float pA0 = __expf(eA0 - MA), pA1 = __expf(eA1 - MA);  // <= 1
    const float pB0 = __expf(eB0 - MB), pB1 = __expf(eB1 - MB);
    sA += pA0 + pA1;
    sB += pB0 + pB1;

    _Float16* Pb = ldsP[(cb >> 6) & 1];
    *(v2h*)(Pb + rA * kPitch + 2 * lane) = v2h{(_Float16)pA0, (_Float16)pA1};
    *(v2h*)(Pb + rB * kPitch + 2 * lane) = v2h{(_Float16)pB0, (_Float16)pB1};
    __syncthreads();   // one barrier per 64 columns (double buffer)

    // ---- two K=32 A fragments from LDS (contiguous K: lane half h holds k 16h..16h+15 of each chunk)
    v16h a0, a1;
    {
      const v8h lo0 = *(const v8h*)(Pb + rl * kPitch + hi * 16);
      const v8h hi0 = *(const v8h*)(Pb + rl * kPitch + 8 + hi * 16);
      const v8h lo1 = *(const v8h*)(Pb + rl * kPitch + 32 + hi * 16);
      const v8h hi1 = *(const v8h*)(Pb + rl * kPitch + 40 + hi * 16);
#pragma unroll
      for (int i = 0; i < 8; ++i) {
        a0[i] = lo0[i]; a0[8 + i] = hi0[i];
        a1[i] = lo1[i]; a1[8 + i] = hi1[i];
      }
    }

    // ---- B fragments from WhT (contiguous 32 B per lane, L2 resident)
    const v16h b00 = *(const v16h*)(bpA + cb);
    const v16h b10 = *(const v16h*)(bpB + cb);
    const v16h b01 = *(const v16h*)(bpA + cb + 32);
    const v16h b11 = *(const v16h*)(bpB + cb + 32);
    acc0 = __builtin_amdgcn_wmma_f32_16x16x32_f16(false, a0, false, b00,
                                                  (short)0, acc0, false, false);
    acc1 = __builtin_amdgcn_wmma_f32_16x16x32_f16(false, a0, false, b10,
                                                  (short)0, acc1, false, false);
    acc0 = __builtin_amdgcn_wmma_f32_16x16x32_f16(false, a1, false, b01,
                                                  (short)0, acc0, false, false);
    acc1 = __builtin_amdgcn_wmma_f32_16x16x32_f16(false, a1, false, b11,
                                                  (short)0, acc1, false, false);
  }

  // ---- finalize: row sums -> LDS, normalize, ELU, store
  sA = wave_sum_f(sA);
  sB = wave_sum_f(sB);
  if (lane == 0) { ldsS[rA] = sA; ldsS[rB] = sB; }
  __syncthreads();
  const v4f s0  = *(const v4f*)(ldsS + hi * 8);
  const v4f s1v = *(const v4f*)(ldsS + hi * 8 + 4);
#pragma unroll
  for (int j = 0; j < 8; ++j) {
    const float sv  = (j < 4) ? s0[j] : s1v[j - 4];
    const float inv = 1.f / sv;
    const int row = r0 + j + hi * 8;
    float o0 = acc0[j] * inv;
    o0 = o0 > 0.f ? o0 : __expf(o0) - 1.f;            // ELU
    out[row * kFout + n0 + rl] = o0;
    float o1 = acc1[j] * inv;
    o1 = o1 > 0.f ? o1 : __expf(o1) - 1.f;
    out[row * kFout + n0 + 16 + rl] = o1;
  }
}

// ---------------------------------------------------------------- launch
extern "C" void kernel_launch(void* const* d_in, const int* in_sizes, int n_in,
                              void* d_out, int out_size, void* d_ws, size_t ws_size,
                              hipStream_t stream) {
  (void)in_sizes; (void)n_in; (void)out_size; (void)ws_size;
  const float* h   = (const float*)d_in[0];   // [8192,512]
  const int*   adj = (const int*)  d_in[1];   // [8192,8192]
  const float* W   = (const float*)d_in[2];   // [512,256]
  const float* av  = (const float*)d_in[3];   // [512,1]
  float* out = (float*)d_out;

  char* ws = (char*)d_ws;
  // workspace layout (all offsets 64B-aligned), ~12.9 MB total
  _Float16* WT    = (_Float16*)(ws);                       // 256*512*2   = 256 KB
  _Float16* WhT   = (_Float16*)(ws + 262144);              // 256*8192*2  = 4 MB
  float*    Wh    = (float*)   (ws + 262144 + 4194304);    // 8192*256*4  = 8 MB
  float*    f1    = (float*)   (ws + 12845056);            // 32 KB
  float*    f2    = (float*)   (ws + 12877824);            // 32 KB
  float*    f2max = (float*)   (ws + 12910592);            // 4 B

  k_transpose_W<<<256,  256, 0, stream>>>(W, WT);
  k_wh         <<<1024, 256, 0, stream>>>(h, WT, Wh, WhT);
  k_f12        <<<1024, 256, 0, stream>>>(Wh, av, f1, f2);
  k_f2max      <<<1,    256, 0, stream>>>(f2, f2max);
  k_attn       <<<512,  256, 0, stream>>>(adj, f1, f2, f2max, WhT, out);
}
